// Encoder_27676769255710
// MI455X (gfx1250) — matrix operand order repaired and checked by reading, not compiled, not run on hardware
//
#include <hip/hip_runtime.h>
#include <hip/hip_bf16.h>
#include <math.h>

#define N_FLOWS 40
#define S 2048
#define D 16384

#define TPB 512                     // 16 waves (wave32)
#define WAVES (TPB / 32)            // 16
#define COLS_PER_WAVE (D / WAVES)   // 1024
#define KITERS (COLS_PER_WAVE / 32) // 32 WMMA steps per wave per layer

typedef __attribute__((ext_vector_type(16))) __bf16 v16bf;
typedef __attribute__((ext_vector_type(8)))  float  v8f;

union PkBf { unsigned u; __bf16 h[2]; };

// ---------------------------------------------------------------------------
// Kernel 1: per-layer parameter prep.
//   wu = w.u ; ww = w.w ; coef = (softplus(wu) - 1 - wu)/ww
//   uhat[l] = u + coef * w           (f32, rank-1 update vector)
//   wpk[l]  = packed bf16 pairs of w (WMMA B operand)
// ---------------------------------------------------------------------------
__global__ __launch_bounds__(256) void prep_kernel(
    const float* __restrict__ ws, const float* __restrict__ us,
    float* __restrict__ uhat, unsigned* __restrict__ wpk)
{
    const int l = blockIdx.x;
    const float* w = ws + (size_t)l * D;
    const float* u = us + (size_t)l * D;

    __shared__ float s_wu[256];
    __shared__ float s_ww[256];
    __shared__ float s_coef;

    float pwu = 0.0f, pww = 0.0f;
    for (int d = threadIdx.x; d < D; d += 256) {
        float wv = w[d];
        pwu += wv * u[d];
        pww += wv * wv;
    }
    s_wu[threadIdx.x] = pwu;
    s_ww[threadIdx.x] = pww;
    __syncthreads();
    for (int off = 128; off > 0; off >>= 1) {
        if (threadIdx.x < off) {
            s_wu[threadIdx.x] += s_wu[threadIdx.x + off];
            s_ww[threadIdx.x] += s_ww[threadIdx.x + off];
        }
        __syncthreads();
    }
    if (threadIdx.x == 0) {
        float wu = s_wu[0], ww = s_ww[0];
        float sp = (wu > 20.0f) ? wu : log1pf(expf(wu));   // softplus
        s_coef = (sp - 1.0f - wu) / ww;
    }
    __syncthreads();
    const float coef = s_coef;

    for (int d = threadIdx.x; d < D; d += 256)
        uhat[(size_t)l * D + d] = u[d] + coef * w[d];

    for (int p = threadIdx.x; p < D / 2; p += 256) {
        PkBf pk;
        pk.h[0] = (__bf16)w[2 * p];
        pk.h[1] = (__bf16)w[2 * p + 1];
        wpk[(size_t)l * (D / 2) + p] = pk.u;
    }
}

// ---------------------------------------------------------------------------
// Kernel 2: fused 40-layer planar-flow chain, 16-row tile per block.
// Layer l pass (thread-private X access):
//   x <- x + act_{l-1}[row]*uhat_{l-1}[col]   (fused update, written back)
//   dot[r] += x_r . w_l  via v_wmma_f32_16x16x32_bf16 (all B columns = w)
// then LDS-reduce 16 dots across waves, act_l = tanh(dot + b_l).
// Final pass applies layer-39's rank-1 update.
// ---------------------------------------------------------------------------
__global__ __launch_bounds__(TPB) void flow_kernel(
    float* __restrict__ X,                 // in-place working copy (d_out)
    const float* __restrict__ uhat_all,
    const unsigned* __restrict__ wpk_all,
    const float* __restrict__ bs)
{
    const int tid  = threadIdx.x;
    const int lane = tid & 31;
    const int wid  = tid >> 5;
    const bool hiH = (lane >= 16);
    const int arow = hiH ? (lane - 16) : lane;      // A-matrix row for this lane

    float* Xt = X + (size_t)blockIdx.x * 16 * D;    // this block's 16-row tile
    float* xrow = Xt + (size_t)arow * D;

    __shared__ float s_part[WAVES * 16];
    __shared__ float s_act[16];

    if (tid < 16) s_act[tid] = 0.0f;
    __syncthreads();

    const int cw   = wid * COLS_PER_WAVE;           // this wave's column span
    const int off0 = hiH ? 8 : 0;

    for (int l = 0; l < N_FLOWS; ++l) {
        const unsigned* wpk = wpk_all + (size_t)l * (D / 2);
        const float* uhp = uhat_all + (size_t)(l > 0 ? l - 1 : 0) * D;
        const bool doUpd = (l > 0);                 // wave-uniform branch
        const float act = s_act[arow];              // act_{l-1} for this row

        v8f c = {0.f, 0.f, 0.f, 0.f, 0.f, 0.f, 0.f, 0.f};
        for (int it = 0; it < KITERS; ++it) {
            const int kb = cw + it * 32;
            // A layout (16-bit A 16x32): lanes 0-15 hold K=0..7 (v0-3) and
            // K=16..23 (v4-7); lanes 16-31 hold K=8..15 and K=24..31.
            const int c0 = kb + off0;
            const int c1 = kb + 16 + off0;
            float4 x0 = *(const float4*)(xrow + c0);
            float4 x1 = *(const float4*)(xrow + c0 + 4);
            float4 x2 = *(const float4*)(xrow + c1);
            float4 x3 = *(const float4*)(xrow + c1 + 4);
            if (doUpd) {   // fused rank-1 update of previous layer
                float4 u0 = *(const float4*)(uhp + c0);
                float4 u1 = *(const float4*)(uhp + c0 + 4);
                float4 u2 = *(const float4*)(uhp + c1);
                float4 u3 = *(const float4*)(uhp + c1 + 4);
                x0.x += act * u0.x; x0.y += act * u0.y; x0.z += act * u0.z; x0.w += act * u0.w;
                x1.x += act * u1.x; x1.y += act * u1.y; x1.z += act * u1.z; x1.w += act * u1.w;
                x2.x += act * u2.x; x2.y += act * u2.y; x2.z += act * u2.z; x2.w += act * u2.w;
                x3.x += act * u3.x; x3.y += act * u3.y; x3.z += act * u3.z; x3.w += act * u3.w;
                *(float4*)(xrow + c0)     = x0;
                *(float4*)(xrow + c0 + 4) = x1;
                *(float4*)(xrow + c1)     = x2;
                *(float4*)(xrow + c1 + 4) = x3;
            }
            v16bf A;
            A[0]=(__bf16)x0.x; A[1]=(__bf16)x0.y; A[2]=(__bf16)x0.z; A[3]=(__bf16)x0.w;
            A[4]=(__bf16)x1.x; A[5]=(__bf16)x1.y; A[6]=(__bf16)x1.z; A[7]=(__bf16)x1.w;
            A[8]=(__bf16)x2.x; A[9]=(__bf16)x2.y; A[10]=(__bf16)x2.z; A[11]=(__bf16)x2.w;
            A[12]=(__bf16)x3.x; A[13]=(__bf16)x3.y; A[14]=(__bf16)x3.z; A[15]=(__bf16)x3.w;
            // B layout (16-bit B 32x16): lanes 0-15 hold K=0..7 and K=16..23,
            // lanes 16-31 hold K=8..15 and K=24..31. Every column j carries the same w chunk, so each
            // accumulator column equals the 16 dot products (exact f32 acc).
            const int pb = (kb >> 1) + (hiH ? 4 : 0);
            v16bf B;
#pragma unroll
            for (int v = 0; v < 8; ++v) {
                PkBf pk; pk.u = wpk[pb + (v < 4 ? v : v + 4)];
                B[2 * v]     = pk.h[0];
                B[2 * v + 1] = pk.h[1];
            }
            c = __builtin_amdgcn_wmma_f32_16x16x32_bf16(
                    false, A, false, B, (short)0, c, false, false);
        }
        // C layout: VGPR r, lane 0 -> (M=r, N=0); lane 16 -> (M=8+r, N=0).
        if (lane == 0) {
#pragma unroll
            for (int r = 0; r < 8; ++r) s_part[wid * 16 + r] = c[r];
        }
        if (lane == 16) {
#pragma unroll
            for (int r = 0; r < 8; ++r) s_part[wid * 16 + 8 + r] = c[r];
        }
        __syncthreads();
        if (tid < 16) {
            float dot = 0.0f;
#pragma unroll
            for (int wv = 0; wv < WAVES; ++wv) dot += s_part[wv * 16 + tid];
            s_act[tid] = tanhf(dot + bs[l]);
        }
        if (l + 1 < N_FLOWS) {   // warm L2/WGP$ for next layer's params
            __builtin_prefetch(uhat_all + (size_t)l * D + tid * 32, 0, 1);
            __builtin_prefetch(wpk_all + (size_t)(l + 1) * (D / 2) + tid * 16, 0, 1);
        }
        __syncthreads();
    }

    // Final pass: apply layer-39's rank-1 update (thread-private columns).
    {
        const float act = s_act[arow];
        const float* uhp = uhat_all + (size_t)(N_FLOWS - 1) * D;
        for (int it = 0; it < KITERS; ++it) {
            const int kb = cw + it * 32;
            const int c0 = kb + off0;
            const int c1 = kb + 16 + off0;
#pragma unroll
            for (int g = 0; g < 2; ++g) {
                const int cc = (g == 0) ? c0 : c1;
                float4 xa = *(const float4*)(xrow + cc);
                float4 xb = *(const float4*)(xrow + cc + 4);
                float4 ua = *(const float4*)(uhp + cc);
                float4 ub = *(const float4*)(uhp + cc + 4);
                xa.x += act * ua.x; xa.y += act * ua.y; xa.z += act * ua.z; xa.w += act * ua.w;
                xb.x += act * ub.x; xb.y += act * ub.y; xb.z += act * ub.z; xb.w += act * ub.w;
                *(float4*)(xrow + cc)     = xa;
                *(float4*)(xrow + cc + 4) = xb;
            }
        }
    }
}

extern "C" void kernel_launch(void* const* d_in, const int* in_sizes, int n_in,
                              void* d_out, int out_size, void* d_ws, size_t ws_size,
                              hipStream_t stream) {
    const float* X  = (const float*)d_in[0];
    const float* ws = (const float*)d_in[1];
    const float* us = (const float*)d_in[2];
    const float* bs = (const float*)d_in[3];
    float* out = (float*)d_out;

    float*    uhat = (float*)d_ws;                                     // 40*D f32
    unsigned* wpk  = (unsigned*)((char*)d_ws +
                                 (size_t)N_FLOWS * D * sizeof(float)); // 40*D/2 u32

    prep_kernel<<<N_FLOWS, 256, 0, stream>>>(ws, us, uhat, wpk);
    hipMemcpyAsync(out, X, (size_t)S * D * sizeof(float),
                   hipMemcpyDeviceToDevice, stream);
    flow_kernel<<<S / 16, TPB, 0, stream>>>(out, uhat, wpk, bs);
}
